// AttentionCritic_12283606466623
// MI455X (gfx1250) — matrix operand order repaired and checked by reading, not compiled, not run on hardware
//
#include <hip/hip_runtime.h>
#include <hip/hip_bf16.h>

// ---------------- problem constants ----------------
constexpr int A_  = 8;        // agents
constexpr int Bb  = 32768;    // batch
constexpr int Ss  = 128;      // state dim
constexpr int ADc = 32;       // action dim
constexpr int Hh  = 128;      // hidden
constexpr int NHh = 4;        // heads
constexpr int Dd  = 32;       // per-head dim
constexpr int SAD = Ss + ADc; // 160
constexpr float SCALE = 0.17677669529663687f; // 1/sqrt(32)

constexpr int TB  = 128;      // batch tile for GEMM kernels
constexpr int TB2 = 32;       // batch tile for attention kernel

// LDS row pads chosen so per-lane strides are bank-conflict-free for b128 loads
constexpr int LDA_IN = SAD + 8;   // 168
constexpr int LDA_H  = Hh + 8;    // 136
constexpr int LDA_2H = 2*Hh + 8;  // 264
constexpr int LDK_H  = Hh + 8;    // 136 (transposed weight row, K=128)
constexpr int LDK_IN = SAD + 8;   // 168 (K=160)
constexpr int LDK_2H = 2*Hh + 8;  // 264 (K=256)

typedef __attribute__((ext_vector_type(16))) _Float16 v16h;
typedef __attribute__((ext_vector_type(8)))  _Float16 v8h;
typedef __attribute__((ext_vector_type(4)))  _Float16 v4h;
typedef __attribute__((ext_vector_type(8)))  float    v8f;
typedef unsigned int u32x4 __attribute__((ext_vector_type(4)));
typedef int          i32x8 __attribute__((ext_vector_type(8)));
typedef int          i32x4 __attribute__((ext_vector_type(4)));

// leaky(x) == fmaxf(x, 0.01x): 2 dual-issueable VALU ops, no cmp/cndmask
__device__ __forceinline__ float leaky(float x) { return fmaxf(x, 0.01f * x); }

// LDS aperture: generic shared-pointer low 32 bits == LDS byte offset (ISA 10.2)
__device__ __forceinline__ unsigned lds_off(const void* p) {
  return (unsigned)(uintptr_t)p;
}

// ---------------- Tensor Data Mover: build D# and issue load ----------------
// 2D/3D tile load, data_size = 4 bytes. x = tile0_dw DWORDs per row,
// y = tile1 rows (stride stride0_dw), z = tile2 planes (stride stride1_dw).
// Optional LDS padding: pad_int_code (0=2..7=256 DWORDs), pad_amt_code (0=1..127=128).
__device__ __forceinline__ void
tdm_load(unsigned ldsoff, unsigned long long gaddr,
         unsigned tile0_dw, unsigned tile1, unsigned tile2,
         unsigned long long stride0_dw, unsigned long long stride1_dw,
         unsigned dim2, int pad_int_code, int pad_amt_code) {
  u32x4 g0;
  g0[0] = 1u;                                   // count=1 (valid), user mode
  g0[1] = ldsoff;                               // lds_addr
  g0[2] = (unsigned)(gaddr & 0xFFFFFFFFull);    // global_addr[95:64]
  g0[3] = (unsigned)((gaddr >> 32) & 0x1FFFFFFull) | (2u << 30); // addr hi | type=2

  unsigned flags = (2u << 16);                  // data_size = 4B; mask=0
  if (pad_amt_code >= 0)
    flags |= (1u << 20) | ((unsigned)pad_int_code << 22) | ((unsigned)pad_amt_code << 25);
  const unsigned td0 = tile0_dw, td1 = tile1;
  i32x8 g1;
  g1[0] = (int)flags;
  g1[1] = (int)((td0 & 0xFFFFu) << 16);                       // tensor_dim0 lo16
  g1[2] = (int)((td0 >> 16) | ((td1 & 0xFFFFu) << 16));       // td0 hi | td1 lo
  g1[3] = (int)((td1 >> 16) | ((tile0_dw & 0xFFFFu) << 16));  // td1 hi | tile_dim0
  g1[4] = (int)((tile1 & 0xFFFFu) | ((tile2 & 0xFFFFu) << 16)); // tile_dim1|2
  g1[5] = (int)(unsigned)(stride0_dw & 0xFFFFFFFFull);        // dim0 stride lo32
  g1[6] = (int)((unsigned)((stride0_dw >> 32) & 0xFFFFull) |
                ((unsigned)(stride1_dw & 0xFFFFull) << 16));  // d0 hi | d1 lo
  g1[7] = (int)(unsigned)((stride1_dw >> 16) & 0xFFFFFFFFull);// dim1 stride hi32
  i32x4 g2; g2[0] = (int)dim2; g2[1] = 0; g2[2] = 0; g2[3] = 0; // tensor_dim2
  i32x4 g3 = {0, 0, 0, 0};
  i32x8 g4 = {0, 0, 0, 0, 0, 0, 0, 0};
  __builtin_amdgcn_tensor_load_to_lds(g0, g1, g2, g3, g4, 0);
}

// ---------------- WMMA fragment loaders (CDNA5 wave32 layouts) ----------------
__device__ __forceinline__ v16h load_a_frag(const _Float16* lds, int lda,
                                            int m0, int k0, int lane) {
  const int m  = m0 + (lane & 15);
  const int h8 = (lane >> 4) << 3;        // 0 or 8
  const _Float16* p = lds + (size_t)m * lda + k0 + h8;
  const v8h lo = *(const v8h*)p;          // K offsets 0..7
  const v8h hi = *(const v8h*)(p + 16);   // K offsets 16..23
  return __builtin_shufflevector(lo, hi, 0,1,2,3,4,5,6,7,8,9,10,11,12,13,14,15);
}

__device__ __forceinline__ v16h load_b_fragT(const _Float16* ldsT, int ldk,
                                             int k0, int n0, int lane) {
  const int n  = n0 + (lane & 15);
  const int kh = k0 + ((lane >> 4) << 3); // +8 for upper half-wave
  const _Float16* p = ldsT + (size_t)n * ldk + kh;
  const v8h lo = *(const v8h*)p;
  const v8h hi = *(const v8h*)(p + 16);
  return __builtin_shufflevector(lo, hi, 0,1,2,3,4,5,6,7,8,9,10,11,12,13,14,15);
}

template <int KT>
__device__ __forceinline__ void preload_a(const _Float16* As, int lda, int m0,
                                          int lane, v16h (&af)[KT]) {
#pragma unroll
  for (int kt = 0; kt < KT; ++kt) af[kt] = load_a_frag(As, lda, m0, kt * 32, lane);
}

template <int KT>
__device__ __forceinline__ v8f gemm_tile(const v16h* af, const _Float16* BsT,
                                         int ldk, int n0, int lane) {
  v8f acc = {};
#pragma unroll
  for (int kt = 0; kt < KT; ++kt) {
    const v16h bf = load_b_fragT(BsT, ldk, kt * 32, n0, lane);
    acc = __builtin_amdgcn_wmma_f32_16x16x32_f16(false, af[kt], false, bf,
                                                 (short)0, acc, false, false);
  }
  return acc;
}

__device__ __forceinline__ void stage_wT(const float* __restrict__ W,
                                         _Float16* wT, int K, int Nn, int ldk,
                                         int tid) {
  for (int idx = tid; idx < K * Nn; idx += 256) {
    const int k = idx / Nn, n = idx % Nn;
    wT[(size_t)n * ldk + k] = (_Float16)W[idx];
  }
}

__device__ __forceinline__ void stage_headWT(const float* __restrict__ Wx,
                                             _Float16* wT, int ldk, int tid) {
  for (int idx = tid; idx < Hh * Hh; idx += 256) {
    const int h = idx / Hh, n = idx % Hh;
    wT[(size_t)n * ldk + h] = (_Float16)Wx[(((n >> 5) * Hh) + h) * Dd + (n & 31)];
  }
}

// ================= K0: BatchNorm statistics =================
__global__ void __launch_bounds__(256)
bn_stats_kernel(const float* __restrict__ states, const float* __restrict__ actions,
                float* __restrict__ mean, float* __restrict__ rstd) {
  const int a   = blockIdx.x / (SAD / 4);
  const int f0  = (blockIdx.x % (SAD / 4)) * 4;
  const int tid = threadIdx.x;

  const float* base; int ld, coff;
  if (f0 < Ss) { base = states  + (size_t)a * Bb * Ss;  ld = Ss;  coff = f0;      }
  else         { base = actions + (size_t)a * Bb * ADc; ld = ADc; coff = f0 - Ss; }

  float4 s = {0,0,0,0}, q = {0,0,0,0};
  for (int b = tid; b < Bb; b += 256) {
    const float4 x = *(const float4*)(base + (size_t)b * ld + coff);
    s.x += x.x; s.y += x.y; s.z += x.z; s.w += x.w;
    q.x += x.x*x.x; q.y += x.y*x.y; q.z += x.z*x.z; q.w += x.w*x.w;
  }
  __shared__ float sh[8][256];
  sh[0][tid]=s.x; sh[1][tid]=s.y; sh[2][tid]=s.z; sh[3][tid]=s.w;
  sh[4][tid]=q.x; sh[5][tid]=q.y; sh[6][tid]=q.z; sh[7][tid]=q.w;
  __syncthreads();
  for (int off = 128; off > 0; off >>= 1) {
    if (tid < off) {
#pragma unroll
      for (int r = 0; r < 8; ++r) sh[r][tid] += sh[r][tid + off];
    }
    __syncthreads();
  }
  if (tid < 4) {
    const float m = sh[tid][0] / (float)Bb;
    const float v = sh[4 + tid][0] / (float)Bb - m * m;
    mean[a * SAD + f0 + tid] = m;
    rstd[a * SAD + f0 + tid] = rsqrtf(v + 1e-5f);
  }
}

// ================= K1: fused encoder + head projections =================
__global__ void __launch_bounds__(256)
encode_kernel(const float* __restrict__ states, const float* __restrict__ actions,
              const float* __restrict__ mean, const float* __restrict__ rstd,
              const float* __restrict__ enc_W, const float* __restrict__ enc_b,
              const float* __restrict__ senc_W, const float* __restrict__ senc_b,
              const float* __restrict__ Wk, const float* __restrict__ Wsel,
              const float* __restrict__ Wv, const float* __restrict__ bv,
              _Float16* __restrict__ s_enc16, _Float16* __restrict__ keys16,
              _Float16* __restrict__ sel16,  _Float16* __restrict__ vals16) {
  const int a    = blockIdx.y;
  const int b0   = blockIdx.x * TB;
  const int tid  = threadIdx.x;
  const int lane = tid & 31;
  const int wave = tid >> 5;
  const int m0   = wave * 16;

  __shared__ _Float16 s_in[TB][LDA_IN];
  __shared__ _Float16 s_wT[Hh][LDK_IN];
  __shared__ _Float16 s_sa[TB][LDA_H];
  __shared__ _Float16 s_se[TB][LDA_H];

  __builtin_prefetch(states + ((size_t)a * Bb + b0 + TB) * Ss, 0, 1);

  const float* mrow = mean + a * SAD;
  const float* rrow = rstd + a * SAD;
  for (int idx = tid; idx < TB * (SAD / 4); idx += 256) {
    const int r  = idx / (SAD / 4);
    const int c4 = (idx % (SAD / 4)) * 4;
    float4 x;
    if (c4 < Ss) x = *(const float4*)&states [((size_t)a * Bb + b0 + r) * Ss  + c4];
    else         x = *(const float4*)&actions[((size_t)a * Bb + b0 + r) * ADc + (c4 - Ss)];
    v4h o;
    o[0] = (_Float16)((x.x - mrow[c4+0]) * rrow[c4+0]);
    o[1] = (_Float16)((x.y - mrow[c4+1]) * rrow[c4+1]);
    o[2] = (_Float16)((x.z - mrow[c4+2]) * rrow[c4+2]);
    o[3] = (_Float16)((x.w - mrow[c4+3]) * rrow[c4+3]);
    *(v4h*)&s_in[r][c4] = o;
  }
  stage_wT(enc_W + (size_t)a * SAD * Hh, &s_wT[0][0], SAD, Hh, LDK_IN, tid);
  __syncthreads();

  v16h af_in[5];
  preload_a<5>(&s_in[0][0], LDA_IN, m0, lane, af_in);

  // ---- GEMM1: sa_enc = leaky(inpn @ enc_W + b), K=160 ----
  for (int nt = 0; nt < 8; ++nt) {
    const v8f acc = gemm_tile<5>(af_in, &s_wT[0][0], LDK_IN, nt * 16, lane);
    const int col   = nt * 16 + (lane & 15);
    const int rbase = m0 + ((lane >> 4) << 3);
    const float bias = enc_b[a * Hh + col];
#pragma unroll
    for (int v = 0; v < 8; ++v)
      s_sa[rbase + v][col] = (_Float16)leaky(acc[v] + bias);
  }
  __syncthreads();
  stage_wT(senc_W + (size_t)a * Ss * Hh, &s_wT[0][0], Ss, Hh, LDK_H, tid);
  __syncthreads();

  // ---- GEMM2: s_enc = leaky(bn(states) @ senc_W + b), K=128 (reuse af_in) ----
  for (int nt = 0; nt < 8; ++nt) {
    const v8f acc = gemm_tile<4>(af_in, &s_wT[0][0], LDK_H, nt * 16, lane);
    const int col   = nt * 16 + (lane & 15);
    const int rbase = m0 + ((lane >> 4) << 3);
    const float bias = senc_b[a * Hh + col];
#pragma unroll
    for (int v = 0; v < 8; ++v) {
      const _Float16 h = (_Float16)leaky(acc[v] + bias);
      s_se[rbase + v][col] = h;
      s_enc16[((size_t)a * Bb + b0 + rbase + v) * Hh + col] = h;
    }
  }
  __syncthreads();

  // ---- GEMM3: keys = sa_enc @ Wk ----
  stage_headWT(Wk, &s_wT[0][0], LDK_H, tid);
  __syncthreads();
  v16h af_sa[4];
  preload_a<4>(&s_sa[0][0], LDA_H, m0, lane, af_sa);
  for (int nt = 0; nt < 8; ++nt) {
    const v8f acc = gemm_tile<4>(af_sa, &s_wT[0][0], LDK_H, nt * 16, lane);
    const int col   = nt * 16 + (lane & 15);
    const int rbase = m0 + ((lane >> 4) << 3);
#pragma unroll
    for (int v = 0; v < 8; ++v)
      keys16[((size_t)a * Bb + b0 + rbase + v) * Hh + col] = (_Float16)acc[v];
  }
  __syncthreads();

  // ---- GEMM4: sel = s_enc @ Wsel ----
  stage_headWT(Wsel, &s_wT[0][0], LDK_H, tid);
  __syncthreads();
  v16h af_se[4];
  preload_a<4>(&s_se[0][0], LDA_H, m0, lane, af_se);
  for (int nt = 0; nt < 8; ++nt) {
    const v8f acc = gemm_tile<4>(af_se, &s_wT[0][0], LDK_H, nt * 16, lane);
    const int col   = nt * 16 + (lane & 15);
    const int rbase = m0 + ((lane >> 4) << 3);
#pragma unroll
    for (int v = 0; v < 8; ++v)
      sel16[((size_t)a * Bb + b0 + rbase + v) * Hh + col] = (_Float16)acc[v];
  }
  __syncthreads();

  // ---- GEMM5: vals = leaky(sa_enc @ Wv + bv) (reuse af_sa) ----
  stage_headWT(Wv, &s_wT[0][0], LDK_H, tid);
  __syncthreads();
  for (int nt = 0; nt < 8; ++nt) {
    const v8f acc = gemm_tile<4>(af_sa, &s_wT[0][0], LDK_H, nt * 16, lane);
    const int col   = nt * 16 + (lane & 15);
    const int rbase = m0 + ((lane >> 4) << 3);
    const float bias = bv[col];
#pragma unroll
    for (int v = 0; v < 8; ++v)
      vals16[((size_t)a * Bb + b0 + rbase + v) * Hh + col] = (_Float16)leaky(acc[v] + bias);
  }
}

// ================= K2: 8x8 cross-agent attention (per head) =================
__global__ void __launch_bounds__(256)
attn_kernel(const _Float16* __restrict__ keys16, const _Float16* __restrict__ sel16,
            const _Float16* __restrict__ vals16, _Float16* __restrict__ other16) {
  const int b0   = blockIdx.x * TB2;
  const int head = blockIdx.y;
  const int tid  = threadIdx.x;

  __shared__ _Float16 sk[A_][TB2][Dd];
  __shared__ _Float16 ss[A_][TB2][Dd];
  __shared__ _Float16 sv[A_][TB2][Dd];

  // TDM: 3D tile per array — x = 16 DWORDs (32 f16), y = 32 rows
  // (stride H/2 DWORDs), z = 8 agents (stride B*H/2 DWORDs). LDS contiguous.
  if (tid < 32) {
    const unsigned long long gelem = (unsigned long long)b0 * Hh + head * Dd;
    const unsigned long long astr  = (unsigned long long)Bb * Hh / 2; // DWORDs
    tdm_load(lds_off(&sk[0][0][0]),
             (unsigned long long)(uintptr_t)(keys16 + gelem),
             Dd / 2, TB2, A_, Hh / 2, astr, A_, -1, -1);
    tdm_load(lds_off(&ss[0][0][0]),
             (unsigned long long)(uintptr_t)(sel16 + gelem),
             Dd / 2, TB2, A_, Hh / 2, astr, A_, -1, -1);
    tdm_load(lds_off(&sv[0][0][0]),
             (unsigned long long)(uintptr_t)(vals16 + gelem),
             Dd / 2, TB2, A_, Hh / 2, astr, A_, -1, -1);
    __builtin_amdgcn_s_wait_tensorcnt(0);
  }
  __syncthreads();

  const int bl = tid >> 3;   // local batch row
  const int i  = tid & 7;    // query agent

  _Float16 selh[Dd];
#pragma unroll
  for (int d8 = 0; d8 < Dd; d8 += 8)
    *(v8h*)&selh[d8] = *(const v8h*)&ss[i][bl][d8];

  float e[A_]; float mx = -1e30f;
#pragma unroll
  for (int j = 0; j < A_; ++j) {
    float lg;
    if (j == i) lg = -1e30f;
    else {
      _Float16 kh[Dd];
#pragma unroll
      for (int d8 = 0; d8 < Dd; d8 += 8)
        *(v8h*)&kh[d8] = *(const v8h*)&sk[j][bl][d8];
      float dot = 0.f;
#pragma unroll
      for (int d = 0; d < Dd; ++d) dot += (float)selh[d] * (float)kh[d];
      lg = dot * SCALE;
    }
    e[j] = lg;
    mx = fmaxf(mx, lg);
  }
  float sum = 0.f;
#pragma unroll
  for (int j = 0; j < A_; ++j) { e[j] = __expf(e[j] - mx); sum += e[j]; }
  const float inv = 1.f / sum;

  float out[Dd];
#pragma unroll
  for (int d = 0; d < Dd; ++d) out[d] = 0.f;
#pragma unroll
  for (int j = 0; j < A_; ++j) {
    const float w = e[j] * inv;
    _Float16 vh[Dd];
#pragma unroll
    for (int d8 = 0; d8 < Dd; d8 += 8)
      *(v8h*)&vh[d8] = *(const v8h*)&sv[j][bl][d8];
#pragma unroll
    for (int d = 0; d < Dd; ++d) out[d] += w * (float)vh[d];
  }
  _Float16 oh[Dd];
#pragma unroll
  for (int d = 0; d < Dd; ++d) oh[d] = (_Float16)out[d];
  _Float16* op = other16 + ((size_t)i * Bb + b0 + bl) * Hh + head * Dd;
#pragma unroll
  for (int d8 = 0; d8 < Dd; d8 += 8)
    *(v8h*)(op + d8) = *(const v8h*)&oh[d8];
}

// ================= K3: critic head + fused argmax gather =================
__global__ void __launch_bounds__(256)
critic_kernel(const _Float16* __restrict__ s_enc16, const _Float16* __restrict__ other16,
              const float* __restrict__ actions,
              const float* __restrict__ c1_W, const float* __restrict__ c1_b,
              const float* __restrict__ c2_W, const float* __restrict__ c2_b,
              float* __restrict__ q_out) {
  const int a    = blockIdx.y;
  const int b0   = blockIdx.x * TB;
  const int tid  = threadIdx.x;
  const int lane = tid & 31;
  const int wave = tid >> 5;
  const int m0   = wave * 16;

  __shared__ _Float16 c_in[TB][LDA_2H];      // [s_enc | other], padded rows
  __shared__ _Float16 w1T[Hh][LDK_2H];
  __shared__ _Float16 h1s[TB][LDA_H];
  __shared__ _Float16 w2T[ADc][LDK_H];
  __shared__ int      ia[TB];

  // TDM: two 2D tiles (64 DWORDs x 128 rows) into interleaved padded rows:
  // pad_interval = 64 DWORDs (code 5), pad_amount = 68 DWORDs (code 67)
  // -> effective LDS row stride 132 DWORDs == LDA_2H f16.
  if (tid < 32) {
    const unsigned long long gelem = ((unsigned long long)a * Bb + b0) * Hh;
    tdm_load(lds_off(&c_in[0][0]),
             (unsigned long long)(uintptr_t)(s_enc16 + gelem),
             Hh / 2, TB, 0, Hh / 2, 0, 0, 5, 67);
    tdm_load(lds_off(&c_in[0][Hh]),
             (unsigned long long)(uintptr_t)(other16 + gelem),
             Hh / 2, TB, 0, Hh / 2, 0, 0, 5, 67);
    __builtin_amdgcn_s_wait_tensorcnt(0);
  }

  // per-row argmax of actions
  for (int r = tid; r < TB; r += 256) {
    const float* ar = actions + ((size_t)a * Bb + b0 + r) * ADc;
    int best = 0; float bvv = ar[0];
    for (int c = 1; c < ADc; ++c) { const float x = ar[c]; if (x > bvv) { bvv = x; best = c; } }
    ia[r] = best;
  }
  stage_wT(c1_W + (size_t)a * 2 * Hh * Hh, &w1T[0][0], 2 * Hh, Hh, LDK_2H, tid);
  stage_wT(c2_W + (size_t)a * Hh * ADc,    &w2T[0][0], Hh, ADc, LDK_H, tid);
  __syncthreads();

  // ---- GEMM1: h1 = leaky(critic_in @ c1_W + b), K=256 ----
  {
    v16h af_ci[8];
    preload_a<8>(&c_in[0][0], LDA_2H, m0, lane, af_ci);
    for (int nt = 0; nt < 8; ++nt) {
      const v8f acc = gemm_tile<8>(af_ci, &w1T[0][0], LDK_2H, nt * 16, lane);
      const int col   = nt * 16 + (lane & 15);
      const int rbase = m0 + ((lane >> 4) << 3);
      const float bias = c1_b[a * Hh + col];
#pragma unroll
      for (int v = 0; v < 8; ++v)
        h1s[rbase + v][col] = (_Float16)leaky(acc[v] + bias);
    }
  }
  __syncthreads();

  // ---- GEMM2: all_q = h1 @ c2_W + b ; store only the argmax column ----
  {
    v16h af_h1[4];
    preload_a<4>(&h1s[0][0], LDA_H, m0, lane, af_h1);
    for (int nt = 0; nt < 2; ++nt) {
      const v8f acc = gemm_tile<4>(af_h1, &w2T[0][0], LDK_H, nt * 16, lane);
      const int col   = nt * 16 + (lane & 15);
      const int rbase = m0 + ((lane >> 4) << 3);
      const float bias = c2_b[a * ADc + col];
#pragma unroll
      for (int v = 0; v < 8; ++v) {
        const int row = rbase + v;
        if (ia[row] == col)
          q_out[(size_t)a * Bb + b0 + row] = acc[v] + bias;
      }
    }
  }
}

// ================= host launcher =================
extern "C" void kernel_launch(void* const* d_in, const int* in_sizes, int n_in,
                              void* d_out, int out_size, void* d_ws, size_t ws_size,
                              hipStream_t stream) {
  (void)in_sizes; (void)n_in; (void)out_size; (void)ws_size;
  const float* states  = (const float*)d_in[0];
  const float* actions = (const float*)d_in[1];
  const float* enc_W   = (const float*)d_in[2];
  const float* enc_b   = (const float*)d_in[3];
  const float* senc_W  = (const float*)d_in[4];
  const float* senc_b  = (const float*)d_in[5];
  const float* Wk      = (const float*)d_in[6];
  const float* Wsel    = (const float*)d_in[7];
  const float* Wv      = (const float*)d_in[8];
  const float* bv      = (const float*)d_in[9];
  const float* c1_W    = (const float*)d_in[10];
  const float* c1_b    = (const float*)d_in[11];
  const float* c2_W    = (const float*)d_in[12];
  const float* c2_b    = (const float*)d_in[13];
  float* q_out = (float*)d_out;

  char* ws = (char*)d_ws;
  float* mean = (float*)ws;
  float* rstd = mean + A_ * SAD;
  size_t off = 16384;
  const size_t planeB = (size_t)A_ * Bb * Hh * sizeof(_Float16); // 64 MB each
  _Float16* s_enc16 = (_Float16*)(ws + off); off += planeB;
  _Float16* keys16  = (_Float16*)(ws + off); off += planeB;
  _Float16* sel16   = (_Float16*)(ws + off); off += planeB;
  _Float16* vals16  = (_Float16*)(ws + off); off += planeB;
  _Float16* other16 = (_Float16*)(ws + off); off += planeB;

  bn_stats_kernel<<<dim3(A_ * (SAD / 4)), 256, 0, stream>>>(states, actions, mean, rstd);

  encode_kernel<<<dim3(Bb / TB, A_), 256, 0, stream>>>(
      states, actions, mean, rstd, enc_W, enc_b, senc_W, senc_b,
      Wk, Wsel, Wv, bv, s_enc16, keys16, sel16, vals16);

  attn_kernel<<<dim3(Bb / TB2, NHh), 256, 0, stream>>>(keys16, sel16, vals16, other16);

  critic_kernel<<<dim3(Bb / TB, A_), 256, 0, stream>>>(
      s_enc16, other16, actions, c1_W, c1_b, c2_W, c2_b, q_out);
}
